// LSTM_56676388438734
// MI455X (gfx1250) — matrix operand order repaired and checked by reading, not compiled, not run on hardware
//
#include <hip/hip_runtime.h>
#include <hip/hip_bf16.h>

#define DIMS  1024
#define VOCAB 32000
#define BATCH 128
#define SEQ   512
#define KTOT  2048          // [x | h] concatenated
#define NTOT  4096          // 1024 units * 4 gates, column order n' = d*4 + g (g: f,i,o,c)
#define KC    64            // K-chunk per staging iteration

typedef __attribute__((ext_vector_type(16))) __bf16 v16bf;
typedef __attribute__((ext_vector_type(8)))  float  v8f;

#if defined(__gfx1250__) && __has_builtin(__builtin_amdgcn_global_load_async_to_lds_b128) && __has_builtin(__builtin_amdgcn_s_wait_asynccnt)
#define LSTM_ASYNC_LDS 1
typedef int v4i __attribute__((vector_size(16)));
typedef __attribute__((address_space(1))) v4i* g_v4i_ptr;   // global
typedef __attribute__((address_space(3))) v4i* l_v4i_ptr;   // LDS
#else
#define LSTM_ASYNC_LDS 0
#endif

static __device__ __forceinline__ unsigned short f2bf(float f) {
    return __builtin_bit_cast(unsigned short, (__bf16)f);   // native cvt, RNE
}
static __device__ __forceinline__ float sigmoidf_(float x) {
    return 1.0f / (1.0f + expf(-x));
}

// 16-byte global -> LDS copy (async on CDNA5, sync fallback otherwise)
static __device__ __forceinline__ void cp16(const unsigned short* g, unsigned short* l) {
#if LSTM_ASYNC_LDS
    __builtin_amdgcn_global_load_async_to_lds_b128((g_v4i_ptr)g, (l_v4i_ptr)l, 0, 0);
#else
    *reinterpret_cast<uint4*>(l) = *reinterpret_cast<const uint4*>(g);
#endif
}
static __device__ __forceinline__ void wait_async_lds() {
#if LSTM_ASYNC_LDS
    __builtin_amdgcn_s_wait_asynccnt(0);
#endif
}

// ---------------------------------------------------------------------------
// Pack weights: Zt[n'][k] (bf16, row-major), n' = d*4 + g, k = 0..2047
//   g = n'&3 (f,i,o,c), d = n'>>2
//   k < 1024 : W_g[k][d]   ;   k >= 1024 : U_g[k-1024][d]
// bias[i] (i < 4096) = b_g[d] with g, d of row n' = i>>11.
// ---------------------------------------------------------------------------
__global__ void lstm_prep_weights(const float* Wf, const float* Uf, const float* bf_,
                                  const float* Wi, const float* Ui, const float* bi_,
                                  const float* Wo, const float* Uo, const float* bo_,
                                  const float* Wc, const float* Uc, const float* bc_,
                                  unsigned short* Zt, float* bias) {
    const float* W[4] = {Wf, Wi, Wo, Wc};
    const float* U[4] = {Uf, Ui, Uo, Uc};
    const float* B[4] = {bf_, bi_, bo_, bc_};
    size_t idx = (size_t)blockIdx.x * blockDim.x + threadIdx.x;
    if (idx >= (size_t)NTOT * KTOT) return;
    int n = (int)(idx >> 11);
    int k = (int)(idx & (KTOT - 1));
    int g = n & 3;
    int d = n >> 2;
    float v = (k < DIMS) ? W[g][(size_t)k * DIMS + d]
                         : U[g][(size_t)(k - DIMS) * DIMS + d];
    Zt[idx] = f2bf(v);
    if (idx < NTOT) bias[idx] = B[g][d];
}

// bf16 copy of the embedding table (one-time).
__global__ void lstm_prep_emb(const float* emb, unsigned short* Ebf) {
    size_t idx = (size_t)blockIdx.x * blockDim.x + threadIdx.x;
    if (idx >= (size_t)VOCAB * DIMS) return;
    Ebf[idx] = f2bf(emb[idx]);
}

// Zero c-state and h-panel buffer 0 (bf16).
__global__ void lstm_init(float* c, unsigned short* H0) {
    int idx = blockIdx.x * blockDim.x + threadIdx.x;
    if (idx >= BATCH * DIMS) return;
    c[idx] = 0.0f;
    H0[idx] = 0;   // bf16(0)
}

// ---------------------------------------------------------------------------
// Fused per-timestep kernel:
//   G = [x_t | h] * Zt^T + bias  (bf16 WMMA, fp32 accum, async double-buffered
//   LDS staging), then gate nonlinearities + c/h update in the epilogue.
// Block: 256 threads (8 waves), tile M=32 x N'=128 (32 units x 4 gates).
// Grid: 4 m-tiles x 32 n-tiles = 128 blocks.
// XMODE 0: A-tiles from pre-converted bf16 tables (pure async copies)
// XMODE 1: x-part converted in-kernel from fp32 embeddings (low-ws fallback)
// ---------------------------------------------------------------------------
#define APAD 72      // halfs per LDS row (64 data + 8 pad)
#define GS_PAD 132   // floats per Gs row

template <int XMODE>
__global__ void lstm_step(const int* __restrict__ features,
                          const float* __restrict__ emb,
                          const unsigned short* __restrict__ Ebf,
                          const unsigned short* __restrict__ Hcur,
                          unsigned short* __restrict__ Hnxt,
                          const unsigned short* __restrict__ Zt,
                          const float* __restrict__ bias,
                          float* __restrict__ c,
                          float* __restrict__ out,
                          int t, int last) {
    __shared__ unsigned short As[2][32 * APAD];     // [buf][m][k]
    __shared__ unsigned short Bs[2][128 * APAD];    // [buf][n][k]
    __shared__ float          Gs[32 * GS_PAD];      // preactivations [m][n_local]

    const int tid  = threadIdx.x;
    const int lane = tid & 31;
    const int w    = tid >> 5;
    const int hi   = (lane >> 4) & 1;
    const int l15  = lane & 15;

    const int mtile = blockIdx.x & 3;          // 4 tiles of 32 batch rows
    const int ntile = blockIdx.x >> 2;         // 32 tiles of 128 gate-cols
    const int m0 = mtile * 32;
    const int n0 = ntile * 128;
    const int wm = (w >> 2) * 16;              // 0 or 16
    const int wn = (w & 3) * 32;               // 0,32,64,96

    // staging coords: A = 32x64 halfs (1 x b128/thread), B = 128x64 (4 x b128)
    const int aRow = tid >> 3;                 // 0..31
    const int aK   = (tid & 7) * 8;            // 0..56
    const int bRow = tid >> 1;                 // 0..127
    const int bK   = (tid & 1) * 32;           // 0 or 32

    const int tok = features[(m0 + aRow) * SEQ + t];
    const unsigned short* ebRow = (XMODE == 0) ? (Ebf + (size_t)tok * DIMS) : nullptr;
    const unsigned short* hRow  = Hcur + (size_t)(m0 + aRow) * DIMS;
    const unsigned short* bRowG = Zt + (size_t)(n0 + bRow) * KTOT;

    auto stage = [&](int kc, int buf) {
        // ---- A tile ----
        if (XMODE == 0) {
            const unsigned short* s = (kc < DIMS) ? (ebRow + kc + aK)
                                                  : (hRow + (kc - DIMS) + aK);
            cp16(s, &As[buf][aRow * APAD + aK]);
        } else {
            if (kc < DIMS) {
                const float* fsrc = &emb[(size_t)tok * DIMS + kc + aK];
                float4 f0 = *reinterpret_cast<const float4*>(fsrc);
                float4 f1 = *reinterpret_cast<const float4*>(fsrc + 4);
                uint4 pk;
                pk.x = (unsigned)f2bf(f0.x) | ((unsigned)f2bf(f0.y) << 16);
                pk.y = (unsigned)f2bf(f0.z) | ((unsigned)f2bf(f0.w) << 16);
                pk.z = (unsigned)f2bf(f1.x) | ((unsigned)f2bf(f1.y) << 16);
                pk.w = (unsigned)f2bf(f1.z) | ((unsigned)f2bf(f1.w) << 16);
                *reinterpret_cast<uint4*>(&As[buf][aRow * APAD + aK]) = pk;
            } else {
                *reinterpret_cast<uint4*>(&As[buf][aRow * APAD + aK]) =
                    *reinterpret_cast<const uint4*>(hRow + (kc - DIMS) + aK);
            }
        }
        // ---- B tile (always async-capable pure copy) ----
        const unsigned short* gs = bRowG + kc + bK;
        unsigned short*       ld = &Bs[buf][bRow * APAD + bK];
        cp16(gs,      ld);
        cp16(gs + 8,  ld + 8);
        cp16(gs + 16, ld + 16);
        cp16(gs + 24, ld + 24);
    };

    // bias folded into accumulator init (bias is column-constant)
    const float bc0 = bias[n0 + wn + l15];
    const float bc1 = bias[n0 + wn + 16 + l15];
    v8f acc0 = {bc0, bc0, bc0, bc0, bc0, bc0, bc0, bc0};
    v8f acc1 = {bc1, bc1, bc1, bc1, bc1, bc1, bc1, bc1};

    stage(0, 0);
    wait_async_lds();
    __syncthreads();

    for (int kc = 0; kc < KTOT; kc += KC) {
        const int cur = (kc >> 6) & 1;
        if (kc + KC < KTOT) stage(kc + KC, cur ^ 1);   // overlap with compute

        #pragma unroll
        for (int sub = 0; sub < 2; ++sub) {
            const int kk = sub * 32;
            // ---- A fragment (16x32, contiguous k = 16*hi + s) ----
            union { v16bf v; unsigned int u[8]; } af;
            const int aOff = (wm + l15) * APAD + kk;
            #pragma unroll
            for (int p = 0; p < 8; ++p) {
                int k0 = 2 * p + hi * 16;
                af.u[p] = *reinterpret_cast<const unsigned int*>(&As[cur][aOff + k0]);
            }
            // ---- B fragments (32x16 each, contiguous k = 16*hi + s) ----
            union { v16bf v; unsigned int u[8]; } bf0, bf1;
            const int bOff0 = (wn + l15) * APAD + kk;
            const int bOff1 = (wn + 16 + l15) * APAD + kk;
            #pragma unroll
            for (int p = 0; p < 8; ++p) {
                int k0 = 2 * p + hi * 16;
                bf0.u[p] = *reinterpret_cast<const unsigned int*>(&Bs[cur][bOff0 + k0]);
                bf1.u[p] = *reinterpret_cast<const unsigned int*>(&Bs[cur][bOff1 + k0]);
            }
            acc0 = __builtin_amdgcn_wmma_f32_16x16x32_bf16(
                       false, af.v, false, bf0.v, (short)0, acc0, false, false);
            acc1 = __builtin_amdgcn_wmma_f32_16x16x32_bf16(
                       false, af.v, false, bf1.v, (short)0, acc1, false, false);
        }

        wait_async_lds();
        __syncthreads();
    }

    // ---- spill preactivations to LDS ----
    {
        const int col0 = wn + l15;
        const int col1 = wn + 16 + l15;
        const int row  = wm + hi * 8;
        #pragma unroll
        for (int r = 0; r < 8; ++r) {
            Gs[(row + r) * GS_PAD + col0] = acc0[r];
            Gs[(row + r) * GS_PAD + col1] = acc1[r];
        }
    }
    __syncthreads();

    // ---- fused gate update: block owns batch rows [m0,m0+32) x units
    //      [d0,d0+32), d0 = n0/4. 1024 (b,d) pairs, 4 per thread. ----
    const int d0 = n0 >> 2;
    #pragma unroll
    for (int i = 0; i < 4; ++i) {
        int p    = tid + i * 256;
        int brow = p >> 5;
        int dloc = p & 31;
        const float* g = &Gs[brow * GS_PAD + dloc * 4];
        float ft = sigmoidf_(g[0]);
        float it = sigmoidf_(g[1]);
        float ot = sigmoidf_(g[2]);
        float ct = tanhf(g[3]);
        int bg = m0 + brow;
        int dg = d0 + dloc;
        size_t cidx = (size_t)bg * DIMS + dg;
        float cn = ft * c[cidx] + it * ct;
        float hn = ot * tanhf(cn);
        c[cidx] = cn;
        Hnxt[cidx] = f2bf(hn);
        if (last) {
            out[(size_t)bg * (2 * DIMS) + dg]        = hn;
            out[(size_t)bg * (2 * DIMS) + DIMS + dg] = cn;
        }
    }
}

// ---------------------------------------------------------------------------
extern "C" void kernel_launch(void* const* d_in, const int* in_sizes, int n_in,
                              void* d_out, int out_size, void* d_ws, size_t ws_size,
                              hipStream_t stream) {
    const int*   features = (const int*)  d_in[0];
    const float* emb      = (const float*)d_in[1];
    const float* Wf = (const float*)d_in[2];
    const float* Uf = (const float*)d_in[3];
    const float* bf_= (const float*)d_in[4];
    const float* Wi = (const float*)d_in[5];
    const float* Ui = (const float*)d_in[6];
    const float* bi_= (const float*)d_in[7];
    const float* Wo = (const float*)d_in[8];
    const float* Uo = (const float*)d_in[9];
    const float* bo_= (const float*)d_in[10];
    const float* Wc = (const float*)d_in[11];
    const float* Uc = (const float*)d_in[12];
    const float* bc_= (const float*)d_in[13];

    char* ws = (char*)d_ws;
    size_t off = 0;
    unsigned short* Zt   = (unsigned short*)(ws + off); off += (size_t)NTOT * KTOT * 2;   // 16 MB
    float*          bias = (float*)(ws + off);          off += (size_t)NTOT * 4;
    unsigned short* H0   = (unsigned short*)(ws + off); off += (size_t)BATCH * DIMS * 2;
    unsigned short* H1   = (unsigned short*)(ws + off); off += (size_t)BATCH * DIMS * 2;
    float*          c    = (float*)(ws + off);          off += (size_t)BATCH * DIMS * 4;
    size_t baseNeed = off;
    unsigned short* Ebf  = (unsigned short*)(ws + off); off += (size_t)VOCAB * DIMS * 2;  // 64 MB
    const bool useEbf = (ws_size >= off);
    (void)baseNeed;

    // 1) pack weights (bf16, gate-interleaved transposed) + bias concat
    {
        size_t total = (size_t)NTOT * KTOT;
        lstm_prep_weights<<<(int)((total + 255) / 256), 256, 0, stream>>>(
            Wf, Uf, bf_, Wi, Ui, bi_, Wo, Uo, bo_, Wc, Uc, bc_, Zt, bias);
    }
    // 1b) bf16 embedding table (only if workspace allows)
    if (useEbf) {
        size_t total = (size_t)VOCAB * DIMS;
        lstm_prep_emb<<<(int)((total + 255) / 256), 256, 0, stream>>>(emb, Ebf);
    }
    // 2) init state
    lstm_init<<<(BATCH * DIMS + 255) / 256, 256, 0, stream>>>(c, H0);

    // 3) recurrence: one fused kernel per timestep
    unsigned short* Hbuf[2] = {H0, H1};
    for (int t = 0; t < SEQ; ++t) {
        if (useEbf) {
            lstm_step<0><<<128, 256, 0, stream>>>(features, emb, Ebf,
                                                  Hbuf[t & 1], Hbuf[(t + 1) & 1],
                                                  Zt, bias, c, (float*)d_out,
                                                  t, (t == SEQ - 1) ? 1 : 0);
        } else {
            lstm_step<1><<<128, 256, 0, stream>>>(features, emb, Ebf,
                                                  Hbuf[t & 1], Hbuf[(t + 1) & 1],
                                                  Zt, bias, c, (float*)d_out,
                                                  t, (t == SEQ - 1) ? 1 : 0);
        }
    }
}
